// SimpleTranslate_62878321213797
// MI455X (gfx1250) — matrix operand order repaired and checked by reading, not compiled, not run on hardware
//
#include <hip/hip_runtime.h>

// ---------------------------------------------------------------------------
// CDNA5 (gfx1250) wave32 WMMA transformer forward + NLL loss.
// All GEMMs use v_wmma_f32_16x16x32_bf16 (bf16 inputs, f32 accumulate).
// All LDS operand tiles are stored so fragment loads are contiguous
// (vectorize to ds_load_b128).
// ---------------------------------------------------------------------------

typedef __attribute__((ext_vector_type(16))) __bf16 v16bf;
typedef __attribute__((ext_vector_type(8)))  float  v8f;

__device__ __forceinline__ __bf16 f2bf(float f) {
  union { float f; unsigned u; } a; a.f = f;
  unsigned r = a.u + 0x7FFFu + ((a.u >> 16) & 1u);   // round-to-nearest-even
  union { unsigned short s; __bf16 b; } o; o.s = (unsigned short)(r >> 16);
  return o.b;
}

__device__ __forceinline__ v8f zero8() {
  v8f z;
#pragma unroll
  for (int i = 0; i < 8; ++i) z[i] = 0.0f;
  return z;
}

__device__ __forceinline__ v8f wmma_bf16(v16bf a, v16bf b, v8f c) {
  // (neg_a, A, neg_b, B, c_mod, C, reuse_a, reuse_b)
  return __builtin_amdgcn_wmma_f32_16x16x32_bf16(false, a, false, b, (short)0, c,
                                                 false, false);
}

// --- fragment loaders (wave32 layouts per CDNA5 ISA 7.12.2) -----------------
// A 16x32 (MxK), row-major LDS [m][k]:
// lane<16 -> M=lane, K in {0..7,16..23}; lane>=16 -> K in {8..15,24..31}
__device__ __forceinline__ v16bf frag_a_bf(const __bf16* base, int stride, int lane) {
  int m = lane & 15, koff = (lane >> 4) * 8;
  v16bf f;
#pragma unroll
  for (int e = 0; e < 16; ++e) {
    int k = koff + e + ((e >= 8) ? 8 : 0);
    f[e] = base[m * stride + k];
  }
  return f;
}
__device__ __forceinline__ v16bf frag_a_f32(const float* base, int stride, int lane) {
  int m = lane & 15, koff = (lane >> 4) * 8;
  v16bf f;
#pragma unroll
  for (int e = 0; e < 16; ++e) {
    int k = koff + e + ((e >= 8) ? 8 : 0);
    f[e] = f2bf(base[m * stride + k]);
  }
  return f;
}
// B 32x16 (KxN) stored TRANSPOSED in LDS ([n][k]); element (k,n) at base[n*stride+k].
// lane<16 -> N=lane, K in {0..7,16..23}; lane>=16 -> K in {8..15,24..31}.  8-runs in k -> b128 loads.
__device__ __forceinline__ v16bf frag_bT_bf(const __bf16* base, int stride, int lane) {
  int n = lane & 15, koff = (lane >> 4) * 8;
  v16bf f;
#pragma unroll
  for (int e = 0; e < 16; ++e) f[e] = base[n * stride + koff + e + ((e >= 8) ? 8 : 0)];
  return f;
}

// ---------------------------------------------------------------------------
__global__ void zero_kernel(float* o) { if (threadIdx.x == 0) o[0] = 0.0f; }

// ---------------------------------------------------------------------------
// embeddings: out[row] = tok_emb[token] + pos_emb[t]; rows = B*256
// ---------------------------------------------------------------------------
__global__ __launch_bounds__(256) void embed_kernel(
    const int* __restrict__ toks, const float* __restrict__ te,
    const float* __restrict__ pe, float* __restrict__ O, int is_dec) {
  int row = blockIdx.x;            // 0..4095
  int b = row >> 8, t = row & 255;
  int tok = is_dec ? ((t < 255) ? toks[b * 256 + t] : 0) : toks[b * 256 + t];
  int tid = threadIdx.x;
  O[row * 512 + tid]       = te[tok * 512 + tid]       + pe[t * 512 + tid];
  O[row * 512 + 256 + tid] = te[tok * 512 + 256 + tid] + pe[t * 512 + 256 + tid];
}

// ---------------------------------------------------------------------------
// LayerNorm over D=512, one block per row
// ---------------------------------------------------------------------------
__global__ __launch_bounds__(256) void ln_kernel(
    const float* __restrict__ X, const float* __restrict__ w,
    const float* __restrict__ b, float* __restrict__ O) {
  __shared__ float red[256];
  int row = blockIdx.x, tid = threadIdx.x;
  const float* xr = X + (size_t)row * 512;
  float v0 = xr[tid], v1 = xr[tid + 256];
  red[tid] = v0 + v1;
  __syncthreads();
  for (int s = 128; s > 0; s >>= 1) {
    if (tid < s) red[tid] += red[tid + s];
    __syncthreads();
  }
  float mu = red[0] * (1.0f / 512.0f);
  __syncthreads();
  float d0 = v0 - mu, d1 = v1 - mu;
  red[tid] = d0 * d0 + d1 * d1;
  __syncthreads();
  for (int s = 128; s > 0; s >>= 1) {
    if (tid < s) red[tid] += red[tid + s];
    __syncthreads();
  }
  float rstd = rsqrtf(red[0] * (1.0f / 512.0f) + 1e-5f);
  O[(size_t)row * 512 + tid]       = d0 * rstd * w[tid]       + b[tid];
  O[(size_t)row * 512 + 256 + tid] = d1 * rstd * w[tid + 256] + b[tid + 256];
}

// ---------------------------------------------------------------------------
// GEMM: C[nrows,N] = A[nrows,K] * B[K,N]  (bf16 WMMA, f32 accum)
// EPI: 0=none  1=+bias+residual  2=gelu(acc+bias)
// BL : 0=row-major B[k*N+n]  1=head-blocked (H,D,64): (n>>6)*K*64 + k*64 + (n&63)
// block 256 threads = 8 waves, 64x64 tile; B staged transposed in LDS
// ---------------------------------------------------------------------------
template <int EPI, int BL>
__global__ __launch_bounds__(256) void gemm_kernel(
    const float* __restrict__ A, const float* __restrict__ Bw,
    const float* __restrict__ bias, const float* __restrict__ res,
    float* __restrict__ C, int nrows, int N, int K) {
  __shared__ __bf16 sA[64 * 40];     // [m][k]
  __shared__ __bf16 sBt[64 * 40];    // [n][k]  (transposed)
  int tid = threadIdx.x, lane = tid & 31, w = tid >> 5;
  int wr = w >> 2, wc = w & 3;
  int row0 = blockIdx.y * 64, col0 = blockIdx.x * 64;
  v8f acc0 = zero8(), acc1 = zero8();
  int nkt = K >> 5;
  for (int kt = 0; kt < nkt; ++kt) {
#pragma unroll
    for (int j = 0; j < 8; ++j) {        // stage A: 64 rows x 32 k
      int i = tid + 256 * j;
      int r = i >> 5, c = i & 31;
      int row = row0 + r;
      float v = (row < nrows) ? A[(size_t)row * K + kt * 32 + c] : 0.0f;
      sA[r * 40 + c] = f2bf(v);
    }
#pragma unroll
    for (int j = 0; j < 8; ++j) {        // stage B: 32 k x 64 n -> [n][k]
      int i = tid + 256 * j;
      int k = i >> 6, n = i & 63;        // n fastest: coalesced global read
      int ng = col0 + n;
      size_t goff = (BL == 1)
        ? (size_t)(ng >> 6) * ((size_t)K * 64) + (size_t)(kt * 32 + k) * 64 + (ng & 63)
        : (size_t)(kt * 32 + k) * N + ng;
      sBt[n * 40 + k] = f2bf(Bw[goff]);
    }
    __syncthreads();
    v16bf bf = frag_bT_bf(sBt + (wc * 16) * 40, 40, lane);
    v16bf a0 = frag_a_bf(sA + (wr * 32) * 40, 40, lane);
    v16bf a1 = frag_a_bf(sA + (wr * 32 + 16) * 40, 40, lane);
    acc0 = wmma_bf16(a0, bf, acc0);
    acc1 = wmma_bf16(a1, bf, acc1);
    __syncthreads();
  }
  int hi = lane >> 4, nl = lane & 15;
  int ncol = col0 + wc * 16 + nl;
#pragma unroll
  for (int r = 0; r < 8; ++r) {
    int m0 = wr * 32 + r + 8 * hi;
#pragma unroll
    for (int half = 0; half < 2; ++half) {
      int row = row0 + m0 + half * 16;
      if (row < nrows) {
        float v = half ? acc1[r] : acc0[r];
        if (EPI == 1) v = v + bias[ncol] + res[(size_t)row * N + ncol];
        if (EPI == 2) {
          v = v + bias[ncol];
          v = 0.5f * v * (1.0f + erff(v * 0.70710678118654752f));
        }
        C[(size_t)row * N + ncol] = v;
      }
    }
  }
}

// ---------------------------------------------------------------------------
// attention: one block per (b, h, 16-query tile); Sq = Skv = 256, DH = 64
// scores = QK^T/8 (WMMA), branchless additive mask, softmax in LDS, out = P V
// block 128 threads = 4 waves
// ---------------------------------------------------------------------------
__global__ __launch_bounds__(128) void attn_kernel(
    const float* __restrict__ Q, const float* __restrict__ Km,
    const float* __restrict__ Vm, float* __restrict__ O,
    const int* __restrict__ src, int causal) {
  __shared__ __bf16 sQ[16 * 72];     // [m][e]
  __shared__ __bf16 sKV[256 * 72];   // K phase: [s][e]; V phase: V^T [e][s] str 264
  __shared__ float  sS[16 * 260];
  __shared__ float  smask[256];
  __shared__ float  redm[16 * 8], reds[16 * 8], rmax[16], rsum[16];
  int b = blockIdx.z, hh = blockIdx.y, t0 = blockIdx.x * 16;
  int tid = threadIdx.x, lane = tid & 31, w = tid >> 5;
  int hi = lane >> 4, nl = lane & 15;
  float cneg = causal ? -1.0e30f : 0.0f;   // uniform causal penalty

#pragma unroll
  for (int j = 0; j < 2; ++j) {          // additive source-pad mask (branchless)
    int s_ = tid + 128 * j;
    int tk = src[b * 256 + s_];
    smask[s_] = (tk == 0 && !causal) ? -1.0e30f : 0.0f;
  }
#pragma unroll
  for (int j = 0; j < 8; ++j) {          // Q tile 16x64
    int i = tid + 128 * j;
    int m = i >> 6, e = i & 63;
    sQ[m * 72 + e] = f2bf(Q[(size_t)(b * 256 + t0 + m) * 512 + hh * 64 + e]);
  }
  for (int j = 0; j < 128; ++j) {        // K 256x64 as [s][e]
    int i = tid + 128 * j;
    int s_ = i >> 6, e = i & 63;
    sKV[s_ * 72 + e] = f2bf(Km[(size_t)(b * 256 + s_) * 512 + hh * 64 + e]);
  }
  __syncthreads();

  int colbase = w * 64;
  v8f acc[4];
#pragma unroll
  for (int nt = 0; nt < 4; ++nt) acc[nt] = zero8();
#pragma unroll
  for (int ks = 0; ks < 2; ++ks) {
    v16bf a = frag_a_bf(sQ + ks * 32, 72, lane);
#pragma unroll
    for (int nt = 0; nt < 4; ++nt) {     // B(k=e,n=s) at sKV[s][e]: transposed form
      v16bf bb = frag_bT_bf(sKV + (colbase + nt * 16) * 72 + ks * 32, 72, lane);
      acc[nt] = wmma_bf16(a, bb, acc[nt]);
    }
  }
#pragma unroll
  for (int nt = 0; nt < 4; ++nt) {
    int sidx = colbase + nt * 16 + nl;
    float madd0 = smask[sidx];           // one LDS read per nt, no branches
#pragma unroll
    for (int r = 0; r < 8; ++r) {
      int m = r + 8 * hi;
      float sc = acc[nt][r] * 0.125f;    // 1/sqrt(64)
      float madd = madd0 + ((sidx > (t0 + m)) ? cneg : 0.0f);
      sS[m * 260 + sidx] = sc + madd;
    }
  }
  __syncthreads();

  // softmax: 8 sub-threads per row
  int rrow = tid & 15, sub = tid >> 4;
  float lmax = -3.0e38f;
  for (int j = 0; j < 32; ++j) lmax = fmaxf(lmax, sS[rrow * 260 + sub * 32 + j]);
  redm[rrow * 8 + sub] = lmax;
  __syncthreads();
  if (tid < 16) {
    float mm = redm[tid * 8];
    for (int j = 1; j < 8; ++j) mm = fmaxf(mm, redm[tid * 8 + j]);
    rmax[tid] = mm;
  }
  __syncthreads();
  float rm = rmax[rrow], lsum = 0.0f;
  for (int j = 0; j < 32; ++j) lsum += __expf(sS[rrow * 260 + sub * 32 + j] - rm);
  reds[rrow * 8 + sub] = lsum;
  __syncthreads();
  if (tid < 16) {
    float ss = 0.0f;
    for (int j = 0; j < 8; ++j) ss += reds[tid * 8 + j];
    rsum[tid] = ss;
  }
  __syncthreads();
  float inv = 1.0f / rsum[rrow];
  for (int j = 0; j < 32; ++j) {
    int c = sub * 32 + j;
    sS[rrow * 260 + c] = __expf(sS[rrow * 260 + c] - rm) * inv;
  }
  for (int j = 0; j < 128; ++j) {        // reload as V^T [e][s], stride 264
    int i = tid + 128 * j;
    int s_ = i >> 6, e = i & 63;         // e fastest: coalesced global read
    sKV[e * 264 + s_] = f2bf(Vm[(size_t)(b * 256 + s_) * 512 + hh * 64 + e]);
  }
  __syncthreads();

  // out tile: each wave 16 cols of DH, K = 256
  v8f oacc = zero8();
#pragma unroll
  for (int ks = 0; ks < 8; ++ks) {
    v16bf a  = frag_a_f32(sS + ks * 32, 260, lane);
    v16bf bb = frag_bT_bf(sKV + (w * 16) * 264 + ks * 32, 264, lane);
    oacc = wmma_bf16(a, bb, oacc);
  }
#pragma unroll
  for (int r = 0; r < 8; ++r) {
    int m = r + 8 * hi;
    O[(size_t)(b * 256 + t0 + m) * 512 + hh * 64 + w * 16 + nl] = oacc[r];
  }
}

// ---------------------------------------------------------------------------
// fused LM head + online log-softmax NLL.  One block per (b, 16-row tile).
// logits tile 16x256 per chunk (4 waves x 64 cols), running (max,sumexp)/row.
// head_w (65MB f32) stays resident in the 192MB L2 across the 256 blocks.
// ---------------------------------------------------------------------------
__global__ __launch_bounds__(128) void loss_kernel(
    const float* __restrict__ Y, const float* __restrict__ HW,
    const float* __restrict__ HB, const int* __restrict__ tdst,
    float* __restrict__ out) {
  __shared__ __bf16 sA[16 * 520];    // [m][k], full K=512 staged once
  __shared__ __bf16 sBt[256 * 40];   // [n][k] per 32-k slab (transposed)
  __shared__ float  sHB[256];
  __shared__ float  srm[4][16], srs[4][16], sT[4][16];
  __shared__ int    stcol[16];
  int b = blockIdx.y, t0 = blockIdx.x * 16;
  int tid = threadIdx.x, lane = tid & 31, w = tid >> 5;
  int hi = lane >> 4, nl = lane & 15;

  for (int j = 0; j < 64; ++j) {         // stage A rows 16x512 once
    int i = tid + 128 * j;
    int m = i >> 9, e = i & 511;
    sA[m * 520 + e] = f2bf(Y[(size_t)(b * 256 + t0 + m) * 512 + e]);
  }
  if (tid < 16) {
    int t = t0 + tid;
    stcol[tid] = (t < 255) ? tdst[b * 256 + t + 1] : -1;
  }
  if (tid < 64) { srm[tid >> 4][tid & 15] = -3.0e38f; srs[tid >> 4][tid & 15] = 0.0f; }
  __syncthreads();

  int tcolr[8];                          // target column per accumulator row
#pragma unroll
  for (int r = 0; r < 8; ++r) tcolr[r] = stcol[r + 8 * hi];
  float tacc[8];
#pragma unroll
  for (int r = 0; r < 8; ++r) tacc[r] = 0.0f;

  for (int ch = 0; ch < 125; ++ch) {     // V = 32000 = 125 * 256
    int col0 = ch * 256;
    if (ch + 1 < 125)                    // nudge next chunk toward L2
      __builtin_prefetch(HW + col0 + 256 + tid, 0, 1);
    __syncthreads();                     // protect sHB vs previous update phase
    sHB[tid]       = HB[col0 + tid];
    sHB[tid + 128] = HB[col0 + tid + 128];
    v8f acc[4];
#pragma unroll
    for (int nt = 0; nt < 4; ++nt) acc[nt] = zero8();
    for (int ks = 0; ks < 16; ++ks) {    // K = 512
      for (int j = 0; j < 64; ++j) {     // stage B 32k x 256n -> [n][k]
        int i = tid + 128 * j;
        int k = i >> 8, n = i & 255;     // n fastest: coalesced global read
        sBt[n * 40 + k] = f2bf(HW[(size_t)(ks * 32 + k) * 32000 + col0 + n]);
      }
      __syncthreads();
      v16bf a = frag_a_bf(sA + ks * 32, 520, lane);
#pragma unroll
      for (int nt = 0; nt < 4; ++nt) {
        v16bf bb = frag_bT_bf(sBt + (w * 64 + nt * 16) * 40, 40, lane);
        acc[nt] = wmma_bf16(a, bb, acc[nt]);
      }
      __syncthreads();
    }
    // per-wave online (max, sumexp) update + target-logit register accumulate
#pragma unroll
    for (int r = 0; r < 8; ++r) {
      int m = r + 8 * hi;
      float vals[4];
      float cm = -3.0e38f;
#pragma unroll
      for (int nt = 0; nt < 4; ++nt) {
        int cloc = w * 64 + nt * 16 + nl;
        float v = acc[nt][r] + sHB[cloc];
        vals[nt] = v;
        cm = fmaxf(cm, v);
        tacc[r] += (col0 + cloc == tcolr[r]) ? v : 0.0f;
      }
      for (int off = 1; off < 16; off <<= 1) cm = fmaxf(cm, __shfl_xor(cm, off, 16));
      float se = 0.0f;
#pragma unroll
      for (int nt = 0; nt < 4; ++nt) se += __expf(vals[nt] - cm);
      for (int off = 1; off < 16; off <<= 1) se += __shfl_xor(se, off, 16);
      if (nl == 0) {
        float M0 = srm[w][m], S0 = srs[w][m];
        float M1 = fmaxf(M0, cm);
        srs[w][m] = S0 * __expf(M0 - M1) + se * __expf(cm - M1);
        srm[w][m] = M1;
      }
    }
  }
  // fold target logit registers: exactly one nonzero contribution per row
#pragma unroll
  for (int r = 0; r < 8; ++r) {
    float tv = tacc[r];
    for (int off = 1; off < 16; off <<= 1) tv += __shfl_xor(tv, off, 16);
    if (nl == 0) sT[w][r + 8 * hi] = tv;
  }
  __syncthreads();
  if (tid < 16) {
    int t = t0 + tid;
    float M = srm[0][tid];
    for (int ww = 1; ww < 4; ++ww) M = fmaxf(M, srm[ww][tid]);
    float S = 0.0f;
    for (int ww = 0; ww < 4; ++ww) S += srs[ww][tid] * __expf(srm[ww][tid] - M);
    float tgt = sT[0][tid] + sT[1][tid] + sT[2][tid] + sT[3][tid];
    float lse = M + __logf(S);
    bool valid = (t < 255) && (tdst[b * 256 + t] != 0);   // PAD == 0
    if (valid) atomicAdd(out, (lse - tgt) * (1.0f / 4080.0f));
  }
}

// ---------------------------------------------------------------------------
// host side
// ---------------------------------------------------------------------------
static void launch_gemm_headed(const float* A, const float* Bw, float* C,
                               hipStream_t s) {
  dim3 g(512 / 64, 4096 / 64);
  gemm_kernel<0, 1><<<g, dim3(256), 0, s>>>(A, Bw, nullptr, nullptr, C, 4096, 512, 512);
}
static void launch_gemm_biasres(const float* A, const float* Bw, const float* bias,
                                const float* res, float* C, int N, int K,
                                hipStream_t s) {
  dim3 g(N / 64, 4096 / 64);
  gemm_kernel<1, 0><<<g, dim3(256), 0, s>>>(A, Bw, bias, res, C, 4096, N, K);
}
static void launch_gemm_gelu(const float* A, const float* Bw, const float* bias,
                             float* C, int N, int K, hipStream_t s) {
  dim3 g(N / 64, 4096 / 64);
  gemm_kernel<2, 0><<<g, dim3(256), 0, s>>>(A, Bw, bias, nullptr, C, 4096, N, K);
}

extern "C" void kernel_launch(void* const* d_in, const int* in_sizes, int n_in,
                              void* d_out, int out_size, void* d_ws, size_t ws_size,
                              hipStream_t stream) {
  (void)in_sizes; (void)n_in; (void)out_size; (void)ws_size;
  int p = 0;
  const int*   tok_src = (const int*)d_in[p++];     // 0
  const int*   tok_dst = (const int*)d_in[p++];     // 1
  const float* tok_emb = (const float*)d_in[p++];   // 2
  const float* pos_emb = (const float*)d_in[p++];   // 3
  // encoder params (insertion order)
  const float* e_ln1w = (const float*)d_in[p++];
  const float* e_ln1b = (const float*)d_in[p++];
  const float* e_wq   = (const float*)d_in[p++];
  const float* e_wk   = (const float*)d_in[p++];
  const float* e_wv   = (const float*)d_in[p++];
  const float* e_wo   = (const float*)d_in[p++];
  const float* e_bo   = (const float*)d_in[p++];
  const float* e_ln2w = (const float*)d_in[p++];
  const float* e_ln2b = (const float*)d_in[p++];
  const float* e_w1   = (const float*)d_in[p++];
  const float* e_b1   = (const float*)d_in[p++];
  const float* e_w2   = (const float*)d_in[p++];
  const float* e_b2   = (const float*)d_in[p++];
  // decoder params
  const float* d_ln1w = (const float*)d_in[p++];
  const float* d_ln1b = (const float*)d_in[p++];
  const float* d_wq   = (const float*)d_in[p++];
  const float* d_wk   = (const float*)d_in[p++];
  const float* d_wv   = (const float*)d_in[p++];
  const float* d_wo   = (const float*)d_in[p++];
  const float* d_bo   = (const float*)d_in[p++];
  const float* d_ln2w = (const float*)d_in[p++];
  const float* d_ln2b = (const float*)d_in[p++];
  const float* d_w1   = (const float*)d_in[p++];
  const float* d_b1   = (const float*)d_in[p++];
  const float* d_w2   = (const float*)d_in[p++];
  const float* d_b2   = (const float*)d_in[p++];
  const float* c_wq   = (const float*)d_in[p++];
  const float* c_wk   = (const float*)d_in[p++];
  const float* c_wv   = (const float*)d_in[p++];
  const float* c_wo   = (const float*)d_in[p++];
  const float* c_bo   = (const float*)d_in[p++];
  const float* d_ln3w = (const float*)d_in[p++];
  const float* d_ln3b = (const float*)d_in[p++];
  const float* ln_ew  = (const float*)d_in[p++];
  const float* ln_eb  = (const float*)d_in[p++];
  const float* ln_dw  = (const float*)d_in[p++];
  const float* ln_db  = (const float*)d_in[p++];
  const float* head_w = (const float*)d_in[p++];
  const float* head_b = (const float*)d_in[p++];

  float* out = (float*)d_out;
  float* ws  = (float*)d_ws;
  const size_t R = 4096;         // B * 256 rows (decoder padded to 256)
  float* x    = ws;              // R*512 each unless noted
  float* y    = x    + R * 512;
  float* h    = y    + R * 512;
  float* qb   = h    + R * 512;
  float* kb   = qb   + R * 512;
  float* vb   = kb   + R * 512;
  float* ab   = vb   + R * 512;
  float* xenc = ab   + R * 512;
  float* ff   = xenc + R * 512;  // R*2048

  const size_t SW = 262144;   // per-layer stride: H*D*DH and (H*DH)*D
  const size_t SF = 1048576;  // per-layer stride: D*M

  zero_kernel<<<1, 32, 0, stream>>>(out);

  // ---------------- encoder ----------------
  embed_kernel<<<4096, 256, 0, stream>>>(tok_src, tok_emb, pos_emb, x, 0);
  for (int i = 0; i < 6; ++i) {
    ln_kernel<<<4096, 256, 0, stream>>>(x, e_ln1w + i * 512, e_ln1b + i * 512, h);
    launch_gemm_headed(h, e_wq + i * SW, qb, stream);
    launch_gemm_headed(h, e_wk + i * SW, kb, stream);
    launch_gemm_headed(h, e_wv + i * SW, vb, stream);
    attn_kernel<<<dim3(16, 8, 16), 128, 0, stream>>>(qb, kb, vb, ab, tok_src, 0);
    launch_gemm_biasres(ab, e_wo + i * SW, e_bo + i * 512, x, x, 512, 512, stream);
    ln_kernel<<<4096, 256, 0, stream>>>(x, e_ln2w + i * 512, e_ln2b + i * 512, h);
    launch_gemm_gelu(h, e_w1 + i * SF, e_b1 + i * 2048, ff, 2048, 512, stream);
    launch_gemm_biasres(ff, e_w2 + i * SF, e_b2 + i * 512, x, x, 512, 2048, stream);
  }
  ln_kernel<<<4096, 256, 0, stream>>>(x, ln_ew, ln_eb, xenc);

  // ---------------- decoder ----------------
  embed_kernel<<<4096, 256, 0, stream>>>(tok_dst, tok_emb, pos_emb, y, 1);
  for (int i = 0; i < 6; ++i) {
    // masked self-attention
    ln_kernel<<<4096, 256, 0, stream>>>(y, d_ln1w + i * 512, d_ln1b + i * 512, h);
    launch_gemm_headed(h, d_wq + i * SW, qb, stream);
    launch_gemm_headed(h, d_wk + i * SW, kb, stream);
    launch_gemm_headed(h, d_wv + i * SW, vb, stream);
    attn_kernel<<<dim3(16, 8, 16), 128, 0, stream>>>(qb, kb, vb, ab, tok_src, 1);
    launch_gemm_biasres(ab, d_wo + i * SW, d_bo + i * 512, y, y, 512, 512, stream);
    // cross-attention (K,V from encoder output)
    ln_kernel<<<4096, 256, 0, stream>>>(y, d_ln2w + i * 512, d_ln2b + i * 512, h);
    launch_gemm_headed(h,    c_wq + i * SW, qb, stream);
    launch_gemm_headed(xenc, c_wk + i * SW, kb, stream);
    launch_gemm_headed(xenc, c_wv + i * SW, vb, stream);
    attn_kernel<<<dim3(16, 8, 16), 128, 0, stream>>>(qb, kb, vb, ab, tok_src, 0);
    launch_gemm_biasres(ab, c_wo + i * SW, c_bo + i * 512, y, y, 512, 512, stream);
    // FFN
    ln_kernel<<<4096, 256, 0, stream>>>(y, d_ln3w + i * 512, d_ln3b + i * 512, h);
    launch_gemm_gelu(h, d_w1 + i * SF, d_b1 + i * 2048, ff, 2048, 512, stream);
    launch_gemm_biasres(ff, d_w2 + i * SF, d_b2 + i * 512, y, y, 512, 2048, stream);
  }
  ln_kernel<<<4096, 256, 0, stream>>>(y, ln_dw, ln_db, h);

  // ---------------- fused head + loss ----------------
  loss_kernel<<<dim3(16, 16), 128, 0, stream>>>(h, head_w, head_b, tok_dst, out);
}
